// VQVAE_30837865185316
// MI455X (gfx1250) — matrix operand order repaired and checked by reading, not compiled, not run on hardware
//
#include <hip/hip_runtime.h>
#include <hip/hip_bf16.h>

// ---------------------------------------------------------------------------
// VQ-VAE forward for MI455X (gfx1250): implicit-GEMM convolutions on
// V_WMMA_F32_16X16X32_BF16 (bf16 inputs, fp32 accumulate), async global->LDS
// weight staging (double buffered), wave32 tiling (8 waves / 256-thread
// block), fused bias/residual epilogues. 32-bit index math throughout.
// ---------------------------------------------------------------------------

typedef __bf16 bf16_t;
typedef bf16_t v16bf __attribute__((ext_vector_type(16)));
typedef float  v8f   __attribute__((ext_vector_type(8)));

__device__ __forceinline__ unsigned short f32_to_bf16(float f) {
  unsigned int u = __float_as_uint(f);
  u = (u + 0x7FFFu + ((u >> 16) & 1u)) >> 16;   // round-to-nearest-even
  return (unsigned short)u;
}
__device__ __forceinline__ float bf16_to_f32(unsigned short h) {
  return __uint_as_float(((unsigned int)h) << 16);
}

union FragBF { int4 q[2]; v16bf v; };
union FragC  { v8f v; float f[8]; };

// ---- async global->LDS (CDNA5) with safe fallback --------------------------
#if defined(__gfx1250__) && __has_builtin(__builtin_amdgcn_global_load_async_to_lds_b128)
#define VQ_USE_ASYNC 1
#else
#define VQ_USE_ASYNC 0
#endif

#if VQ_USE_ASYNC
typedef int v4i_vec __attribute__((vector_size(4 * sizeof(int))));   // clang vector, not HIP int4
typedef __attribute__((address_space(1))) v4i_vec gas_v4i;           // global (AS1)
typedef __attribute__((address_space(3))) v4i_vec las_v4i;           // LDS    (AS3)
#endif

__device__ __forceinline__ void lds_copy_b128(const unsigned short* g, unsigned short* l) {
#if VQ_USE_ASYNC
  gas_v4i* gp = (gas_v4i*)(size_t)g;                     // int->ptr: AS1 typed
  las_v4i* lp = (las_v4i*)(unsigned)(size_t)l;           // low 32 bits = LDS offset
  __builtin_amdgcn_global_load_async_to_lds_b128(gp, lp, 0, 0);
#else
  *reinterpret_cast<int4*>(l) = *reinterpret_cast<const int4*>(g);
#endif
}

__device__ __forceinline__ void wait_async_le(int more) {
#if VQ_USE_ASYNC
#if __has_builtin(__builtin_amdgcn_s_wait_asynccnt)
  if (more) __builtin_amdgcn_s_wait_asynccnt(1);
  else      __builtin_amdgcn_s_wait_asynccnt(0);
#else
  if (more) __asm__ volatile("s_wait_asynccnt 0x1" ::: "memory");
  else      __asm__ volatile("s_wait_asynccnt 0x0" ::: "memory");
#endif
#else
  (void)more;
#endif
}

// ---------------------------------------------------------------------------
// Implicit-GEMM conv:  C[M=B*Ho*Wo, N=Cout] = A[M, K=KH*KW*Cin] * B[K, N]
// A built on the fly (im2col) from NHWC bf16 activations; B = prepped bf16
// weights [Npad][Kpad] row-major. Block tile M=128 x N=64, K-step 32.
// 8 waves: wm 0..3 x wn 0..1, each wave a 32x32 tile via 2x2 WMMA fragments.
// ---------------------------------------------------------------------------
#define TM 128
#define TN 64
#define TK 32

__global__ __launch_bounds__(256)
void conv_ig_kernel(const unsigned short* __restrict__ Xb,
                    const unsigned short* __restrict__ Wt,
                    const float* __restrict__ bias,
                    unsigned short* __restrict__ Yb,
                    float* __restrict__ Yf,
                    const unsigned short* __restrict__ Res,
                    int Bn, int Hi, int Wi, int Cin,
                    int Ho, int Wo, int Cout,
                    int KH, int KW, int stride, int pad,
                    int Ktot, int Kpad)
{
  __shared__ __align__(16) unsigned short As[TM * TK];        // 8 KB
  __shared__ __align__(16) unsigned short Bs[2][TN * TK];     // 8 KB (dbl buf)

  const int tid  = threadIdx.x;
  const int lane = tid & 31;
  const int wave = tid >> 5;
  const int wm   = wave & 3;
  const int wn   = wave >> 2;
  const int m0   = blockIdx.x * TM;
  const int n0   = blockIdx.y * TN;
  const int Mtot = Bn * Ho * Wo;

  // ---- per-thread fixed A assignment: row aml, 16 consecutive K at akk ----
  const int aml = tid >> 1;                 // 0..127
  const int akk = (tid & 1) << 4;           // 0 or 16
  const int am  = m0 + aml;
  const bool mvalid = (am < Mtot);
  int t2 = am / Wo;
  const int aox = am - t2 * Wo;
  const int aoy = t2 % Ho;
  const int abb = t2 / Ho;
  const int oybase = aoy * stride - pad;
  const int oxbase = aox * stride - pad;
  const unsigned pixbase = (unsigned)abb * (unsigned)(Hi * Wi * Cin);

  // ---- per-thread fixed B assignment: one 16B segment ----
  const int bidx = tid * 8;                 // 0..2047
  const int bnl  = bidx >> 5;
  const int bkk  = bidx & 31;
  const unsigned brow = (unsigned)(n0 + bnl) * (unsigned)Kpad + bkk;

  FragC acc[2][2];
  for (int i = 0; i < 2; ++i)
    for (int j = 0; j < 2; ++j)
      for (int r = 0; r < 8; ++r) acc[i][j].f[r] = 0.0f;

  const int al = lane & 15;
  const int arow_off = (lane < 16) ? 0 : 8;    // A lane K {0..7,16..23} / {8..15,24..31}
  const int brow_off = (lane < 16) ? 0 : 8;    // B lane K {0..7,16..23} / {8..15,24..31}

  // prologue: stage B tile for k0 = 0
  lds_copy_b128(&Wt[brow], &Bs[0][bidx]);
  int buf = 0;

  for (int k0 = 0; k0 < Kpad; k0 += TK) {
    // ---- stage A tile: build 16 bf16 in registers, two ds_store_b128 ----
    {
      int k = k0 + akk;
      int ci = 0, kx = 0, ky = 0;
      if (k < Ktot) { ci = k % Cin; int t = k / Cin; kx = t % KW; ky = t / KW; }
      union { unsigned short u[16]; int4 q[2]; } tmp;
#pragma unroll
      for (int e = 0; e < 16; ++e) {
        unsigned short v = 0;
        if (mvalid && (k + e) < Ktot) {
          int iy = oybase + ky;
          int ix = oxbase + kx;
          if (iy >= 0 && iy < Hi && ix >= 0 && ix < Wi)
            v = Xb[pixbase + (unsigned)(iy * Wi + ix) * (unsigned)Cin + (unsigned)ci];
        }
        tmp.u[e] = v;
        if (++ci == Cin) { ci = 0; if (++kx == KW) { kx = 0; ++ky; } }
      }
      int4* d4 = reinterpret_cast<int4*>(&As[aml * TK + akk]);
      d4[0] = tmp.q[0];
      d4[1] = tmp.q[1];
    }

    // ---- issue async stage of next B tile into the other buffer ----
    const int more = (k0 + TK < Kpad);
    if (more) {
      lds_copy_b128(&Wt[brow + (unsigned)(k0 + TK)], &Bs[buf ^ 1][bidx]);
      __builtin_prefetch(&Wt[brow + (unsigned)(k0 + 2 * TK)], 0, 1);
    }
    wait_async_le(more);   // current tile resident (count <= in-flight next tile)
    __syncthreads();

    // ---- fragments per ISA 7.12.2 layouts ----
    FragBF a[2], b[2];
#pragma unroll
    for (int i = 0; i < 2; ++i) {
      const unsigned short* pa = &As[(wm * 32 + i * 16 + al) * TK];
      a[i].q[0] = *reinterpret_cast<const int4*>(pa + arow_off);
      a[i].q[1] = *reinterpret_cast<const int4*>(pa + arow_off + 16);
    }
#pragma unroll
    for (int j = 0; j < 2; ++j) {
      const unsigned short* pb = &Bs[buf][(wn * 32 + j * 16 + al) * TK];
      b[j].q[0] = *reinterpret_cast<const int4*>(pb + brow_off);
      b[j].q[1] = *reinterpret_cast<const int4*>(pb + brow_off + 16);
    }
#pragma unroll
    for (int i = 0; i < 2; ++i)
#pragma unroll
      for (int j = 0; j < 2; ++j)
        acc[i][j].v = __builtin_amdgcn_wmma_f32_16x16x32_bf16(
            false, a[i].v, false, b[j].v, (short)0, acc[i][j].v, false, false);
    __syncthreads();
    buf ^= 1;
  }

  // ---- epilogue: bias + residual + store (bf16 or fp32), 32-bit offsets ----
#pragma unroll
  for (int i = 0; i < 2; ++i)
#pragma unroll
    for (int j = 0; j < 2; ++j)
#pragma unroll
      for (int r = 0; r < 8; ++r) {
        int ml = wm * 32 + i * 16 + r + ((lane >> 4) << 3);
        int nl = wn * 32 + j * 16 + (lane & 15);
        int m = m0 + ml;
        int n = n0 + nl;
        if (m < Mtot && n < Cout) {
          float v = acc[i][j].f[r];
          if (bias) v += bias[n];
          unsigned o = (unsigned)m * (unsigned)Cout + (unsigned)n;
          if (Res) v += bf16_to_f32(Res[o]);
          if (Yf) Yf[o] = v;
          else    Yb[o] = f32_to_bf16(v);
        }
      }
}

// OIHW fp32 -> [Npad][Kpad] bf16, K = (ky*KW+kx)*Cin + cin, zero padded.
__global__ __launch_bounds__(256)
void prep_w_kernel(const float* __restrict__ w, unsigned short* __restrict__ wt,
                   int Cout, int Cin, int KH, int KW, int Ktot, int Kpad, int Npad)
{
  int idx = blockIdx.x * 256 + threadIdx.x;
  if (idx >= Npad * Kpad) return;
  int n = idx / Kpad;
  int k = idx - n * Kpad;
  float v = 0.0f;
  if (n < Cout && k < Ktot) {
    int ci = k % Cin; int t = k / Cin;
    int kx = t % KW;  int ky = t / KW;
    v = w[((n * Cin + ci) * KH + ky) * KW + kx];
  }
  wt[idx] = f32_to_bf16(v);
}

// GroupNorm(32 groups) + swish, NHWC bf16 -> bf16. One block per (batch,group).
// Cg = C/32 is a power of two -> shift/mask indexing.
__global__ __launch_bounds__(256)
void gn_swish_kernel(const unsigned short* __restrict__ X,
                     const float* __restrict__ w, const float* __restrict__ b,
                     unsigned short* __restrict__ Y,
                     int C, int HW, int cgsh)
{
  const int n = blockIdx.x >> 5;            // 32 groups
  const int g = blockIdx.x & 31;
  const int Cg = 1 << cgsh;
  const int cmask = Cg - 1;
  const int count = HW << cgsh;
  const int tid = threadIdx.x;
  const unsigned base = (unsigned)n * (unsigned)HW * (unsigned)C + (unsigned)(g << cgsh);
  __shared__ float s1[256], s2[256];

  float sum = 0.0f, sq = 0.0f;
  for (int i = tid; i < count; i += 256) {
    int s = i >> cgsh;
    int c = i & cmask;
    float v = bf16_to_f32(X[base + (unsigned)s * (unsigned)C + (unsigned)c]);
    sum += v; sq += v * v;
  }
  s1[tid] = sum; s2[tid] = sq;
  __syncthreads();
  for (int s = 128; s > 0; s >>= 1) {
    if (tid < s) { s1[tid] += s1[tid + s]; s2[tid] += s2[tid + s]; }
    __syncthreads();
  }
  float inv = 1.0f / (float)count;
  float mu  = s1[0] * inv;
  float var = s2[0] * inv - mu * mu;
  float rstd = rsqrtf(var + 1e-5f);

  for (int i = tid; i < count; i += 256) {
    int s = i >> cgsh;
    int c = i & cmask;
    unsigned a = base + (unsigned)s * (unsigned)C + (unsigned)c;
    int ch = (g << cgsh) + c;
    float v = bf16_to_f32(X[a]);
    float y = (v - mu) * rstd * w[ch] + b[ch];
    y = y / (1.0f + __expf(-y));            // swish always follows GN here
    Y[a] = f32_to_bf16(y);
  }
}

// Vector quantization: one block per z-vector; forward value = codebook[argmin].
__global__ __launch_bounds__(256)
void vq_kernel(const unsigned short* __restrict__ ze,
               const float* __restrict__ codebook,
               unsigned short* __restrict__ zq,
               int D, int K)
{
  const int vec = blockIdx.x;
  const int tid = threadIdx.x;
  __shared__ float sz[256];
  __shared__ float sd[256];
  __shared__ int   si[256];

  for (int d = tid; d < D; d += 256)
    sz[d] = bf16_to_f32(ze[(unsigned)vec * (unsigned)D + d]);
  __syncthreads();

  float best = 3.4e38f; int bi = 0;
  for (int k = tid; k < K; k += 256) {
    const float* cb = &codebook[(unsigned)k * (unsigned)D];
    float dist = 0.0f;
    for (int d = 0; d < D; ++d) {
      float df = sz[d] - cb[d];
      dist += df * df;
    }
    if (dist < best) { best = dist; bi = k; }
  }
  sd[tid] = best; si[tid] = bi;
  __syncthreads();
  for (int s = 128; s > 0; s >>= 1) {
    if (tid < s && sd[tid + s] < sd[tid]) { sd[tid] = sd[tid + s]; si[tid] = si[tid + s]; }
    __syncthreads();
  }
  int idx = si[0];
  for (int d = tid; d < D; d += 256)
    zq[(unsigned)vec * (unsigned)D + d] = f32_to_bf16(codebook[(unsigned)idx * (unsigned)D + d]);
}

// ConvTranspose2d k=4 s=2 p=1 (cin==cout==C), NHWC bf16, fp32 weights OIHW.
__global__ __launch_bounds__(256)
void convt_kernel(const unsigned short* __restrict__ X,
                  const float* __restrict__ Wq, const float* __restrict__ bias,
                  unsigned short* __restrict__ Y,
                  int Bn, int Hi, int Wi, int C)
{
  const int Ho = Hi * 2, Wo = Wi * 2;
  int idx = blockIdx.x * 256 + threadIdx.x;
  int total = Bn * Ho * Wo * C;
  if (idx >= total) return;
  int co = idx % C; int t = idx / C;
  int ox = t % Wo; t /= Wo;
  int oy = t % Ho; int bb = t / Ho;

  float acc = bias[co];
  for (int ky = 0; ky < 4; ++ky) {
    int jy = oy + ky - 2;
    if (jy < 0 || (jy & 1) || (jy >> 1) >= Hi) continue;
    int iy = jy >> 1;
    for (int kx = 0; kx < 4; ++kx) {
      int jx = ox + kx - 2;
      if (jx < 0 || (jx & 1) || (jx >> 1) >= Wi) continue;
      int ix = jx >> 1;
      const unsigned short* xp = &X[(unsigned)((bb * Hi + iy) * Wi + ix) * (unsigned)C];
      const float* wp = &Wq[(unsigned)co * (unsigned)C * 16u + (unsigned)(ky * 4 + kx)];
      for (int ci = 0; ci < C; ++ci)
        acc += bf16_to_f32(xp[ci]) * wp[(unsigned)ci * 16u];
    }
  }
  Y[idx] = f32_to_bf16(acc);
}

__global__ __launch_bounds__(256)
void cvt_f32_bf16_kernel(const float* __restrict__ x, unsigned short* __restrict__ y, int n)
{
  int i = blockIdx.x * 256 + threadIdx.x;
  if (i < n) y[i] = f32_to_bf16(x[i]);
}

// ---------------------------------------------------------------------------
// Host orchestration
// ---------------------------------------------------------------------------
struct ResP { const float *c1, *c2, *c3, *n1b, *n1w, *n2b, *n2w; int cin, cout; };

static inline long cdivl(long a, long b) { return (a + b - 1) / b; }
static inline int ilog2i(int v) { int s = 0; while ((1 << s) < v) ++s; return s; }

static void run_gn_swish(hipStream_t st, const unsigned short* x, const float* w,
                         const float* b, unsigned short* y, int Bn, int C, int HW)
{
  gn_swish_kernel<<<dim3(Bn * 32), 256, 0, st>>>(x, w, b, y, C, HW, ilog2i(C / 32));
}

static void run_conv(hipStream_t st, unsigned short* wslab,
                     const float* w, const float* bias,
                     const unsigned short* src, unsigned short* dst, float* dstf,
                     const unsigned short* res,
                     int Bn, int H, int W, int Cin, int Cout,
                     int KH, int stride, int pad)
{
  int Ktot = KH * KH * Cin;
  int Kpad = ((Ktot + 31) / 32) * 32;
  int Npad = ((Cout + 63) / 64) * 64;
  long wtElems = (long)Npad * Kpad;
  prep_w_kernel<<<dim3((unsigned)cdivl(wtElems, 256)), 256, 0, st>>>(
      w, wslab, Cout, Cin, KH, KH, Ktot, Kpad, Npad);
  int Ho = (H + 2 * pad - KH) / stride + 1;
  int Wo = (W + 2 * pad - KH) / stride + 1;
  long M = (long)Bn * Ho * Wo;
  dim3 grid((unsigned)cdivl(M, TM), (unsigned)(Npad / TN));
  conv_ig_kernel<<<grid, 256, 0, st>>>(src, wslab, bias, dst, dstf, res,
                                       Bn, H, W, Cin, Ho, Wo, Cout,
                                       KH, KH, stride, pad, Ktot, Kpad);
}

static void run_resblock(hipStream_t st, unsigned short* wslab, const ResP& r,
                         unsigned short* in, unsigned short* out,
                         unsigned short* T1, unsigned short* T2,
                         int Bn, int H, int W)
{
  run_gn_swish(st, in, r.n1w, r.n1b, T1, Bn, r.cin, H * W);
  run_conv(st, wslab, r.c1, nullptr, T1, T2, nullptr, nullptr,
           Bn, H, W, r.cin, r.cout, 3, 1, 1);
  run_gn_swish(st, T2, r.n2w, r.n2b, T1, Bn, r.cout, H * W);
  const unsigned short* res = in;
  if (r.c3) {
    run_conv(st, wslab, r.c3, nullptr, in, out, nullptr, nullptr,
             Bn, H, W, r.cin, r.cout, 1, 1, 0);
    res = out;   // in-place residual read/write is per-element safe
  }
  run_conv(st, wslab, r.c2, nullptr, T1, out, nullptr, res,
           Bn, H, W, r.cout, r.cout, 3, 1, 1);
}

extern "C" void kernel_launch(void* const* d_in, const int* in_sizes, int n_in,
                              void* d_out, int out_size, void* d_ws, size_t ws_size,
                              hipStream_t stream)
{
  (void)in_sizes; (void)n_in; (void)out_size; (void)ws_size;

  // ---- walk jax pytree flattening: dict keys alphabetical, lists in order ----
  int ip = 0;
  auto F = [&]() -> const float* { return (const float*)d_in[ip++]; };
  auto read_res = [&](int cin, int cout) -> ResP {
    ResP r; r.cin = cin; r.cout = cout;
    r.c1 = F(); r.c2 = F();
    r.c3 = (cin != cout) ? F() : nullptr;
    r.n1b = F(); r.n1w = F(); r.n2b = F(); r.n2w = F();
    return r;
  };

  const float* x_in = F();                                    // x [4,128,128,3]

  // encoder: conv_in_b, conv_in_w, conv_out_b, conv_out_w, down[...], mid1, mid2, nob, now
  const float* e_cin_b = F();  const float* e_cin_w = F();
  const float* e_cout_b = F(); const float* e_cout_w = F();
  const int ech[5] = {128, 128, 256, 256, 512};
  const float *e_down_b[4], *e_down_w[4];
  ResP e_res[5][2];
  int curr = 128;
  for (int i = 0; i < 5; ++i) {
    if (i < 4) { e_down_b[i] = F(); e_down_w[i] = F(); }      // 'down_b' < 'down_w' < 'res'
    for (int rI = 0; rI < 2; ++rI) { e_res[i][rI] = read_res(curr, ech[i]); curr = ech[i]; }
  }
  ResP e_mid1 = read_res(512, 512);
  ResP e_mid2 = read_res(512, 512);
  const float* e_nob = F(); const float* e_now = F();

  // decoder: conv_in_b/w, conv_out_b/w, mid1, mid2, nob, now, up[ res..., up_b, up_w ]
  const float* d_cin_b = F();  const float* d_cin_w = F();
  const float* d_cout_b = F(); const float* d_cout_w = F();
  ResP d_mid1 = read_res(512, 512);
  ResP d_mid2 = read_res(512, 512);
  const float* d_nob = F(); const float* d_now = F();
  const int dch[5] = {512, 256, 256, 128, 128};
  const float *d_up_b[4], *d_up_w[4];
  ResP d_res[5][2];
  curr = 512;
  for (int i = 0; i < 5; ++i) {
    for (int rI = 0; rI < 2; ++rI) { d_res[i][rI] = read_res(curr, dch[i]); curr = dch[i]; }
    if (i < 4) { d_up_b[i] = F(); d_up_w[i] = F(); }          // 'res' < 'up_b' < 'up_w'
  }
  const float* codebook = F();

  // ---- workspace layout: weight slab + 4 bf16 activation slabs ----
  char* ws = (char*)d_ws;
  unsigned short* WS = (unsigned short*)ws;                   // 8 MB weight slab
  const size_t SLAB = (size_t)1 << 24;                        // 16.78 MB each
  unsigned short* S0 = (unsigned short*)(ws + ((size_t)8 << 20));
  unsigned short* S1 = (unsigned short*)(ws + ((size_t)8 << 20) + SLAB);
  unsigned short* S2 = (unsigned short*)(ws + ((size_t)8 << 20) + 2 * SLAB);
  unsigned short* S3 = (unsigned short*)(ws + ((size_t)8 << 20) + 3 * SLAB);

  const int B = 4;
  hipStream_t st = stream;

  // ---- encoder ----
  {
    int n = B * 128 * 128 * 3;
    cvt_f32_bf16_kernel<<<dim3((unsigned)cdivl(n, 256)), 256, 0, st>>>(x_in, S0, n);
  }
  unsigned short* cur = S1; unsigned short* alt = S0;
  run_conv(st, WS, e_cin_w, e_cin_b, S0, S1, nullptr, nullptr, B, 128, 128, 3, 128, 3, 1, 1);
  int H = 128;
  for (int i = 0; i < 5; ++i) {
    for (int rI = 0; rI < 2; ++rI) {
      run_resblock(st, WS, e_res[i][rI], cur, alt, S2, S3, B, H, H);
      unsigned short* t = cur; cur = alt; alt = t;
    }
    if (i < 4) {
      run_conv(st, WS, e_down_w[i], e_down_b[i], cur, alt, nullptr, nullptr,
               B, H, H, ech[i], ech[i], 3, 2, 1);
      unsigned short* t = cur; cur = alt; alt = t;
      H /= 2;
    }
  }
  run_resblock(st, WS, e_mid1, cur, alt, S2, S3, B, H, H); { unsigned short* t=cur; cur=alt; alt=t; }
  run_resblock(st, WS, e_mid2, cur, alt, S2, S3, B, H, H); { unsigned short* t=cur; cur=alt; alt=t; }
  run_gn_swish(st, cur, e_now, e_nob, alt, B, 512, H * H);
  run_conv(st, WS, e_cout_w, e_cout_b, alt, cur, nullptr, nullptr, B, H, H, 512, 256, 3, 1, 1);
  // cur = z_e bf16 [4,8,8,256]

  // ---- vector quantization (forward value == z_q) ----
  vq_kernel<<<dim3(B * 8 * 8), 256, 0, st>>>(cur, codebook, alt, 256, 512);
  { unsigned short* t = cur; cur = alt; alt = t; }            // cur = z_q

  // ---- decoder ----
  run_conv(st, WS, d_cin_w, d_cin_b, cur, alt, nullptr, nullptr, B, H, H, 256, 512, 3, 1, 1);
  { unsigned short* t = cur; cur = alt; alt = t; }
  run_resblock(st, WS, d_mid1, cur, alt, S2, S3, B, H, H); { unsigned short* t=cur; cur=alt; alt=t; }
  run_resblock(st, WS, d_mid2, cur, alt, S2, S3, B, H, H); { unsigned short* t=cur; cur=alt; alt=t; }
  for (int i = 0; i < 5; ++i) {
    for (int rI = 0; rI < 2; ++rI) {
      run_resblock(st, WS, d_res[i][rI], cur, alt, S2, S3, B, H, H);
      unsigned short* t = cur; cur = alt; alt = t;
    }
    if (i < 4) {
      long tot = (long)B * (2 * H) * (2 * H) * dch[i];
      convt_kernel<<<dim3((unsigned)cdivl(tot, 256)), 256, 0, st>>>(
          cur, d_up_w[i], d_up_b[i], alt, B, H, H, dch[i]);
      unsigned short* t = cur; cur = alt; alt = t;
      H *= 2;
    }
  }
  run_gn_swish(st, cur, d_now, d_nob, alt, B, 128, H * H);
  run_conv(st, WS, d_cout_w, d_cout_b, alt, nullptr, (float*)d_out, nullptr,
           B, H, H, 128, 3, 3, 1, 1);                         // fp32 NHWC output
}
